// LinearAttention1d_6786048328223
// MI455X (gfx1250) — matrix operand order repaired and checked by reading, not compiled, not run on hardware
//
#include <hip/hip_runtime.h>
#include <hip/hip_bf16.h>

// ---------------------------------------------------------------------------
// LinearAttention1d for MI455X (gfx1250, wave32, WMMA 16x16x32 f16->f32)
// B=8, C=512, L=4096, H=8, D=64, hidden=512, 3*hidden=1536
// ---------------------------------------------------------------------------

typedef __attribute__((ext_vector_type(16))) _Float16 v16h;
typedef __attribute__((ext_vector_type(8)))  float    v8f;

constexpr int BB  = 8;
constexpr int CC  = 512;
constexpr int LL  = 4096;
constexpr int HH  = 8;
constexpr int DHD = 64;
constexpr int HID = 512;
constexpr int O3  = 1536;

#define SCALE_Q   0.125f         // DIM_HEAD^-0.5
#define LN_EPS    1e-5f
#define CTX_UP    1024.0f        // rescale context -> f16 normal range
#define ATT_UP2   64.0f          // extra rescale at attn store (total 2^16)
#define ATT_DOWN  (1.0f/65536.0f)

__device__ inline v8f zero8() {
  v8f z;
#pragma unroll
  for (int e = 0; e < 8; ++e) z[e] = 0.0f;
  return z;
}

__device__ inline v8f wmma_f16(v16h a, v16h b, v8f c) {
  return __builtin_amdgcn_wmma_f32_16x16x32_f16(
      /*neg_a=*/false, a, /*neg_b=*/false, b,
      /*c_mod=*/(short)0, c, /*reuse_a=*/false, /*reuse_b=*/false);
}

// A fragment (16x32, f16) from row-major f32 [M x K], at (m0, k0).
// lanes 0-15: M=m0+lane, K in {k0..k0+7, k0+16..k0+23}
// lanes 16-31: same M, K in {k0+8..k0+15, k0+24..k0+31}
__device__ inline v16h loadA_f32(const float* __restrict__ A, int ld,
                                 int m0, int k0, int lane) {
  int row = m0 + (lane & 15);
  int kb  = k0 + ((lane >> 4) << 3);
  const float* p = A + (size_t)row * ld + kb;
  v16h f;
#pragma unroll
  for (int e = 0; e < 8; ++e) { f[e] = (_Float16)p[e]; f[e + 8] = (_Float16)p[e + 16]; }
  return f;
}

// A fragment from row-major f16 [M x K] (pure b128 loads, no cvt)
__device__ inline v16h loadA_f16(const _Float16* __restrict__ A, int ld,
                                 int m0, int k0, int lane) {
  int row = m0 + (lane & 15);
  int kb  = k0 + ((lane >> 4) << 4);
  const _Float16* p = A + (size_t)row * ld + kb;
  v16h f;
#pragma unroll
  for (int e = 0; e < 8; ++e) { f[e] = p[e]; f[e + 8] = p[e + 8]; }
  return f;
}

// B fragment (32x16, f16) from row-major f32 [K x N], at (k0, n0).
// lanes 0-15: N=n0+lane, K=k0..k0+15 ; lanes 16-31: K=k0+16..k0+31
__device__ inline v16h loadB_f32(const float* __restrict__ Bm, int ld,
                                 int k0, int n0, int lane) {
  int n = n0 + (lane & 15);
  const float* p = Bm + (size_t)(k0 + ((lane >> 4) << 4)) * ld + n;
  v16h f;
#pragma unroll
  for (int e = 0; e < 16; ++e) f[e] = (_Float16)p[(size_t)e * ld];
  return f;
}

// B fragment from row-major f16 [K x N]
__device__ inline v16h loadB_f16(const _Float16* __restrict__ Bm, int ld,
                                 int k0, int n0, int lane) {
  int n = n0 + (lane & 15);
  const _Float16* p = Bm + (size_t)(k0 + ((lane >> 4) << 4)) * ld + n;
  v16h f;
#pragma unroll
  for (int e = 0; e < 16; ++e) f[e] = p[(size_t)e * ld];
  return f;
}

// ---------------------------------------------------------------------------
// K0: one-time f32 -> f16 conversion of weight matrices into workspace
// ---------------------------------------------------------------------------
__global__ __launch_bounds__(256) void k_cvt_w(const float* __restrict__ src,
                                               _Float16* __restrict__ dst,
                                               int n) {
  int i = (blockIdx.x * 256 + threadIdx.x) * 4;
  if (i + 3 < n) {
#pragma unroll
    for (int e = 0; e < 4; ++e) dst[i + e] = (_Float16)src[i + e];
  }
}

// ---------------------------------------------------------------------------
// K1: qkv[b] = W_qkv[1536x512] @ X[b][512x4096]  -> f16 workspace
// Block tile 128x128, 8 waves (2 in M x 4 in N), wave tile 64x32.
// ---------------------------------------------------------------------------
__global__ __launch_bounds__(256) void k_qkv_gemm(
    const float* __restrict__ x, const _Float16* __restrict__ wq16,
    _Float16* __restrict__ qkv16) {
  int lane = threadIdx.x & 31, w = threadIdx.x >> 5;
  int b  = blockIdx.z;
  int m0 = blockIdx.y * 128 + (w & 1) * 64;
  int n0 = blockIdx.x * 128 + (w >> 1) * 32;
  const float* Xb = x + (size_t)b * CC * LL;

  v8f acc[4][2];
#pragma unroll
  for (int i = 0; i < 4; ++i)
#pragma unroll
    for (int j = 0; j < 2; ++j) acc[i][j] = zero8();

  for (int kk = 0; kk < CC; kk += 32) {
    if (kk + 32 < CC) {  // prefetch next K-tile of the streaming operand
      __builtin_prefetch(Xb + (size_t)(kk + 32 + (lane >> 4) * 16) * LL +
                             n0 + (lane & 15) * 4, 0, 1);
    }
    v16h a[4], bf[2];
#pragma unroll
    for (int i = 0; i < 4; ++i) a[i] = loadA_f16(wq16, CC, m0 + 16 * i, kk, lane);
#pragma unroll
    for (int j = 0; j < 2; ++j) bf[j] = loadB_f32(Xb, LL, kk, n0 + 16 * j, lane);
#pragma unroll
    for (int i = 0; i < 4; ++i)
#pragma unroll
      for (int j = 0; j < 2; ++j) acc[i][j] = wmma_f16(a[i], bf[j], acc[i][j]);
  }

  _Float16* ob = qkv16 + (size_t)b * O3 * LL;
  int half8 = (lane >> 4) << 3, c15 = lane & 15;
#pragma unroll
  for (int i = 0; i < 4; ++i)
#pragma unroll
    for (int j = 0; j < 2; ++j)
#pragma unroll
      for (int r = 0; r < 8; ++r) {
        int row = m0 + 16 * i + r + half8;
        int col = n0 + 16 * j + c15;
        ob[(size_t)row * LL + col] = (_Float16)acc[i][j][r];
      }
}

// ---------------------------------------------------------------------------
// K2a: per (b,h,d) row of k: max and 1/sum(exp) over L=4096
// ---------------------------------------------------------------------------
__global__ __launch_bounds__(256) void k_kstats(
    const _Float16* __restrict__ qkv16,
    float* __restrict__ kmax, float* __restrict__ kinv) {
  int rid = blockIdx.x;                 // b*512 + (h*64+d)
  int b = rid >> 9, o = rid & 511;
  const _Float16* row = qkv16 + ((size_t)b * O3 + HID + o) * LL;
  __shared__ float sred[256];
  int tid = threadIdx.x;

  float m = -1e30f;
  for (int i = tid; i < LL; i += 256) m = fmaxf(m, (float)row[i]);
  sred[tid] = m; __syncthreads();
  for (int s = 128; s; s >>= 1) {
    if (tid < s) sred[tid] = fmaxf(sred[tid], sred[tid + s]);
    __syncthreads();
  }
  float mx = sred[0]; __syncthreads();

  float ss = 0.0f;
  for (int i = tid; i < LL; i += 256) ss += __expf((float)row[i] - mx);
  sred[tid] = ss; __syncthreads();
  for (int s = 128; s; s >>= 1) {
    if (tid < s) sred[tid] += sred[tid + s];
    __syncthreads();
  }
  if (tid == 0) { kmax[rid] = mx; kinv[rid] = 1.0f / sred[0]; }
}

// ---------------------------------------------------------------------------
// K2b: context[b,h][64x64] = softmax_n(k)[64x4096] @ (v/L)^T[4096x64]
// One block per (b,h); 8 waves cover 4x4 16x16 tiles (2 e-tiles each).
// exp transform applied in the A-fragment loader.
// ---------------------------------------------------------------------------
__global__ __launch_bounds__(256) void k_context(
    const _Float16* __restrict__ qkv16,
    const float* __restrict__ kmax, const float* __restrict__ kinv,
    float* __restrict__ ctx) {
  int bh = blockIdx.x;                  // b*8+h
  int b = bh >> 3, h = bh & 7;
  int lane = threadIdx.x & 31, w = threadIdx.x >> 5;
  int dT = w & 3, e0 = (w >> 2) * 2;
  int m0 = dT * 16;
  const _Float16* kb_ = qkv16 + ((size_t)b * O3 + HID + h * DHD) * LL;
  const _Float16* vb_ = qkv16 + ((size_t)b * O3 + 2 * HID + h * DHD) * LL;

  int mrow = m0 + (lane & 15);
  float mx  = kmax[bh * DHD + mrow];
  float inv = kinv[bh * DHD + mrow];

  v8f acc[2];
  acc[0] = zero8(); acc[1] = zero8();

  for (int k0 = 0; k0 < LL; k0 += 32) {
    if (k0 + 32 < LL) {
      __builtin_prefetch(kb_ + (size_t)mrow * LL + k0 + 32, 0, 1);
    }
    const _Float16* pa = kb_ + (size_t)mrow * LL + k0 + ((lane >> 4) << 4);
    v16h a;
#pragma unroll
    for (int e = 0; e < 8; ++e) {
      a[e]     = (_Float16)(__expf((float)pa[e]      - mx) * inv);
      a[e + 8] = (_Float16)(__expf((float)pa[e + 8] - mx) * inv);
    }
#pragma unroll
    for (int j = 0; j < 2; ++j) {
      int col = (e0 + j) * 16 + (lane & 15);         // e index
      const _Float16* pb = vb_ + (size_t)col * LL + k0 + ((lane >> 4) << 4);
      v16h bf;
#pragma unroll
      for (int e = 0; e < 16; ++e) bf[e] = (_Float16)((float)pb[e] * (1.0f / LL));
      acc[j] = wmma_f16(a, bf, acc[j]);
    }
  }

  int half8 = (lane >> 4) << 3, c15 = lane & 15;
#pragma unroll
  for (int j = 0; j < 2; ++j)
#pragma unroll
    for (int r = 0; r < 8; ++r)
      ctx[((size_t)bh * DHD + m0 + r + half8) * DHD + (e0 + j) * 16 + c15] =
          acc[j][r];
}

// ---------------------------------------------------------------------------
// K3a: in-place softmax over d (64) of q columns, times SCALE, f16
// ---------------------------------------------------------------------------
__global__ __launch_bounds__(256) void k_qsoftmax(_Float16* __restrict__ qkv16) {
  int b = blockIdx.z, h = blockIdx.y;
  int n = blockIdx.x * 256 + threadIdx.x;
  _Float16* base = qkv16 + ((size_t)b * O3 + h * DHD) * LL + n;
  float vals[DHD];
  float mx = -1e30f;
#pragma unroll
  for (int d = 0; d < DHD; ++d) {
    vals[d] = (float)base[(size_t)d * LL];
    mx = fmaxf(mx, vals[d]);
  }
  float s = 0.0f;
#pragma unroll
  for (int d = 0; d < DHD; ++d) { vals[d] = __expf(vals[d] - mx); s += vals[d]; }
  float inv = SCALE_Q / s;
#pragma unroll
  for (int d = 0; d < DHD; ++d) base[(size_t)d * LL] = (_Float16)(vals[d] * inv);
}

// ---------------------------------------------------------------------------
// K3b: attn[b][h*64+e][n] = sum_d ctx[d][e] * q_soft[d][n]   (x 2^16 rescale)
// Block: (b,h) x 128-column slab. 8 waves: 4 mTiles x 8 nTiles.
// ---------------------------------------------------------------------------
__global__ __launch_bounds__(256) void k_attn(
    const float* __restrict__ ctx, const _Float16* __restrict__ qkv16,
    _Float16* __restrict__ attn16) {
  int bh = blockIdx.y; int b = bh >> 3, h = bh & 7;
  int lane = threadIdx.x & 31, w = threadIdx.x >> 5;
  int mT = w & 3, nT0 = (w >> 2) * 4;
  int m0 = mT * 16;
  int n0 = blockIdx.x * 128 + nT0 * 16;
  const float*    cb = ctx + (size_t)bh * DHD * DHD;    // [d][e]
  const _Float16* qb = qkv16 + ((size_t)b * O3 + h * DHD) * LL;

  v8f acc[4];
#pragma unroll
  for (int j = 0; j < 4; ++j) acc[j] = zero8();

  int m = m0 + (lane & 15);                              // e row
  for (int k0 = 0; k0 < DHD; k0 += 32) {
    int kb = k0 + ((lane >> 4) << 4);
    v16h a;                                              // A = ctx^T (e x d)
#pragma unroll
    for (int e = 0; e < 8; ++e) {
      a[e]     = (_Float16)(cb[(size_t)(kb + e) * DHD + m] * CTX_UP);
      a[e + 8] = (_Float16)(cb[(size_t)(kb + 8 + e) * DHD + m] * CTX_UP);
    }
#pragma unroll
    for (int j = 0; j < 4; ++j) {
      v16h bf = loadB_f16(qb, LL, k0, n0 + 16 * j, lane);
      acc[j] = wmma_f16(a, bf, acc[j]);
    }
  }

  _Float16* ob = attn16 + (size_t)b * HID * LL;
  int half8 = (lane >> 4) << 3, c15 = lane & 15;
#pragma unroll
  for (int j = 0; j < 4; ++j)
#pragma unroll
    for (int r = 0; r < 8; ++r) {
      int row = h * DHD + m0 + r + half8;
      int col = n0 + 16 * j + c15;
      ob[(size_t)row * LL + col] = (_Float16)(acc[j][r] * ATT_UP2);
    }
}

// ---------------------------------------------------------------------------
// K4: proj = W_out @ attn * 2^-16 + b_out, then LayerNorm over 512 channels,
// scale by g. One block owns the full 512-channel column slab of 64 l's,
// so LN is done with one LDS cross-wave reduction (no extra pass).
// ---------------------------------------------------------------------------
__global__ __launch_bounds__(256) void k_out_ln(
    const _Float16* __restrict__ attn16, const _Float16* __restrict__ wo16,
    const float* __restrict__ b_out, const float* __restrict__ g,
    float* __restrict__ out) {
  int b  = blockIdx.y;
  int l0 = blockIdx.x * 64;
  int lane = threadIdx.x & 31, w = threadIdx.x >> 5;
  int wm = w * 64;                                       // wave's 64-row strip
  const _Float16* ab = attn16 + (size_t)b * HID * LL;

  v8f acc[4][4];
#pragma unroll
  for (int i = 0; i < 4; ++i)
#pragma unroll
    for (int j = 0; j < 4; ++j) acc[i][j] = zero8();

  for (int kk = 0; kk < HID; kk += 32) {
    if (kk + 32 < HID) {
      __builtin_prefetch(ab + (size_t)(kk + 32 + (lane >> 4) * 16) * LL +
                             l0 + (lane & 15) * 4, 0, 1);
    }
    v16h a[4], bf[4];
#pragma unroll
    for (int i = 0; i < 4; ++i) a[i] = loadA_f16(wo16, HID, wm + 16 * i, kk, lane);
#pragma unroll
    for (int j = 0; j < 4; ++j) bf[j] = loadB_f16(ab, LL, kk, l0 + 16 * j, lane);
#pragma unroll
    for (int i = 0; i < 4; ++i)
#pragma unroll
      for (int j = 0; j < 4; ++j) acc[i][j] = wmma_f16(a[i], bf[j], acc[i][j]);
  }

  int half8 = (lane >> 4) << 3, c15 = lane & 15;
  // undo rescale, add bias
#pragma unroll
  for (int i = 0; i < 4; ++i)
#pragma unroll
    for (int j = 0; j < 4; ++j)
#pragma unroll
      for (int r = 0; r < 8; ++r) {
        int row = wm + 16 * i + r + half8;
        acc[i][j][r] = acc[i][j][r] * ATT_DOWN + b_out[row];
      }

  __shared__ float s_s[8][32][4];
  __shared__ float s_q[8][32][4];
  __shared__ float s_mean[64];
  __shared__ float s_inv[64];

#pragma unroll
  for (int j = 0; j < 4; ++j) {
    float ls = 0.0f, lq = 0.0f;
#pragma unroll
    for (int i = 0; i < 4; ++i)
#pragma unroll
      for (int r = 0; r < 8; ++r) {
        float v = acc[i][j][r];
        ls += v; lq += v * v;
      }
    s_s[w][lane][j] = ls;
    s_q[w][lane][j] = lq;
  }
  __syncthreads();

  int tid = threadIdx.x;
  if (tid < 64) {
    int j = tid >> 4, ln = tid & 15;
    float ts = 0.0f, tq = 0.0f;
#pragma unroll
    for (int ww = 0; ww < 8; ++ww) {
      ts += s_s[ww][ln][j] + s_s[ww][ln + 16][j];
      tq += s_q[ww][ln][j] + s_q[ww][ln + 16][j];
    }
    float mean = ts * (1.0f / 512.0f);
    float var  = tq * (1.0f / 512.0f) - mean * mean;
    s_mean[tid] = mean;
    s_inv[tid]  = rsqrtf(var + LN_EPS);
  }
  __syncthreads();

  float* outb = out + (size_t)b * CC * LL;
#pragma unroll
  for (int i = 0; i < 4; ++i)
#pragma unroll
    for (int j = 0; j < 4; ++j)
#pragma unroll
      for (int r = 0; r < 8; ++r) {
        int row  = wm + 16 * i + r + half8;
        int colj = 16 * j + c15;
        float v = (acc[i][j][r] - s_mean[colj]) * s_inv[colj] * g[row];
        outb[(size_t)row * LL + (l0 + colj)] = v;
      }
}

// ---------------------------------------------------------------------------
extern "C" void kernel_launch(void* const* d_in, const int* in_sizes, int n_in,
                              void* d_out, int out_size, void* d_ws,
                              size_t ws_size, hipStream_t stream) {
  (void)in_sizes; (void)n_in; (void)out_size; (void)ws_size;
  const float* x     = (const float*)d_in[0];
  const float* w_qkv = (const float*)d_in[1];
  const float* w_out = (const float*)d_in[2];
  const float* b_out = (const float*)d_in[3];
  const float* g     = (const float*)d_in[4];
  float* out = (float*)d_out;

  char* ws = (char*)d_ws;
  size_t off = 0;
  _Float16* qkv16 = (_Float16*)(ws + off); off += (size_t)BB * O3 * LL * sizeof(_Float16);
  float* kmax     = (float*)(ws + off);    off += (size_t)BB * HID * sizeof(float);
  float* kinv     = (float*)(ws + off);    off += (size_t)BB * HID * sizeof(float);
  float* ctx      = (float*)(ws + off);    off += (size_t)BB * HH * DHD * DHD * sizeof(float);
  _Float16* attn16 = (_Float16*)(ws + off); off += (size_t)BB * HID * LL * sizeof(_Float16);
  _Float16* wq16  = (_Float16*)(ws + off); off += (size_t)O3 * CC * sizeof(_Float16);
  _Float16* wo16  = (_Float16*)(ws + off);

  // one-time weight conversion to f16 (removes per-block cvt work in GEMMs)
  k_cvt_w<<<dim3((O3 * CC) / 1024), 256, 0, stream>>>(w_qkv, wq16, O3 * CC);
  k_cvt_w<<<dim3((CC * HID) / 1024), 256, 0, stream>>>(w_out, wo16, CC * HID);

  k_qkv_gemm<<<dim3(LL / 128, O3 / 128, BB), 256, 0, stream>>>(x, wq16, qkv16);
  k_kstats<<<dim3(BB * HID), 256, 0, stream>>>(qkv16, kmax, kinv);
  k_context<<<dim3(BB * HH), 256, 0, stream>>>(qkv16, kmax, kinv, ctx);
  k_qsoftmax<<<dim3(LL / 256, HH, BB), 256, 0, stream>>>(qkv16);
  k_attn<<<dim3(LL / 128, BB * HH), 256, 0, stream>>>(ctx, qkv16, attn16);
  k_out_ln<<<dim3(LL / 64, BB), 256, 0, stream>>>(attn16, wo16, b_out, g, out);
}
